// QModule_64415919506102
// MI455X (gfx1250) — 8-bit matrix operand order repaired and checked by reading, compiled, not run on hardware
//
#include <hip/hip_runtime.h>
#include <stdint.h>


typedef __attribute__((ext_vector_type(8))) int          v8i;
typedef __attribute__((ext_vector_type(4))) int          v4i;
typedef __attribute__((ext_vector_type(2))) int          v2i;
typedef __attribute__((ext_vector_type(4))) unsigned int v4u;

#define M_DIM 8192
#define N_DIM 4096
#define K_DIM 1024

#define KSTEP   64
#define TILE_MN 128
#define LDS_ROW 80

__device__ __forceinline__ void tdm_load_tile_128x64(const int8_t* gptr,
                                                     uint32_t lds_off) {
    const uint64_t ga = (uint64_t)(uintptr_t)gptr;
    v4u g0;
    g0[0] = 1u;
    g0[1] = lds_off;
    g0[2] = (uint32_t)(ga & 0xffffffffu);
    g0[3] = (uint32_t)((ga >> 32) & 0x01ffffffu)
          | (2u << 30);

    v8i g1;
    g1[0] = (int)((1u << 20)
                | (3u << 22)
                | (3u << 25));
    g1[1] = (int)((KSTEP & 0xffffu) << 16);
    g1[2] = (int)(((KSTEP >> 16) & 0xffffu)
                | ((TILE_MN & 0xffffu) << 16));
    g1[3] = (int)(((TILE_MN >> 16) & 0xffffu)
                | ((unsigned)KSTEP << 16));
    g1[4] = (int)(TILE_MN);
    g1[5] = (int)(K_DIM);
    g1[6] = 0;
    g1[7] = 0;

    const v4i z4 = {0, 0, 0, 0};
    const v8i z8 = {0, 0, 0, 0, 0, 0, 0, 0};
    __builtin_amdgcn_tensor_load_to_lds(g0, g1, z4, z4, z8,  0);
}

__global__ __launch_bounds__(256) void int8_gemm_wmma_tdm(
    const int8_t* __restrict__ X,
    const int8_t* __restrict__ Wq,
    const float*  __restrict__ scale_i,
    const float*  __restrict__ scale_w,
    const float*  __restrict__ bias,
    float*        __restrict__ out)
{
    __shared__ __align__(16) int8_t sA[2][TILE_MN * LDS_ROW];
    __shared__ __align__(16) int8_t sB[2][TILE_MN * LDS_ROW];

    const int lane   = threadIdx.x & 31;
    const int wave   = threadIdx.x >> 5;
    const int wave_m = wave >> 2;
    const int wave_n = wave & 3;

    const int half = lane >> 4;
    const int l16  = lane & 15;

    const int m_base = blockIdx.y * TILE_MN + wave_m * 64;
    const int n_base = blockIdx.x * TILE_MN + wave_n * 32;

    const int8_t* Atile = X  + (size_t)(blockIdx.y * TILE_MN) * K_DIM;
    const int8_t* Btile = Wq + (size_t)(blockIdx.x * TILE_MN) * K_DIM;

    const int a_off = (wave_m * 64 + l16) * LDS_ROW + half * 16;
    const int b_off = (wave_n * 32 + l16) * LDS_ROW + half * 16;

    v8i acc[4][2];
#pragma unroll
    for (int mi = 0; mi < 4; ++mi)
#pragma unroll
        for (int ni = 0; ni < 2; ++ni)
            acc[mi][ni] = (v8i){0, 0, 0, 0, 0, 0, 0, 0};

    if (wave == 0) {
        tdm_load_tile_128x64(Atile, (uint32_t)(uintptr_t)&sA[0][0]);
        tdm_load_tile_128x64(Btile, (uint32_t)(uintptr_t)&sB[0][0]);
    }

#pragma unroll 2
    for (int kt = 0; kt < K_DIM / KSTEP; ++kt) {
        const int cur = kt & 1;

        if (wave == 0) __builtin_amdgcn_s_wait_tensorcnt(0);
        __syncthreads();

        if (wave == 0 && kt + 1 < K_DIM / KSTEP) {
            const size_t knext = (size_t)(kt + 1) * KSTEP;
            tdm_load_tile_128x64(Atile + knext,
                                 (uint32_t)(uintptr_t)&sA[cur ^ 1][0]);
            tdm_load_tile_128x64(Btile + knext,
                                 (uint32_t)(uintptr_t)&sB[cur ^ 1][0]);
        }

        v8i afrag[4];
#pragma unroll
        for (int mi = 0; mi < 4; ++mi) {
            const int8_t* p = &sA[cur][0] + a_off + mi * 16 * LDS_ROW;
            v8i a;
#pragma unroll
            for (int j = 0; j < 4; ++j) {
                v2i d = *(const v2i*)(p + 8 * (j & 1) + 32 * (j >> 1));
                a[2 * j]     = d[0];
                a[2 * j + 1] = d[1];
            }
            afrag[mi] = a;
        }

        v8i bfrag[2];
#pragma unroll
        for (int ni = 0; ni < 2; ++ni) {
            const int8_t* p  = &sB[cur][0] + b_off + ni * 16 * LDS_ROW;
            v4i d0 = *(const v4i*)(p);
            v4i d1 = *(const v4i*)(p + 32);
            v8i b;
            b[0] = d0[0]; b[1] = d0[1]; b[2] = d0[2]; b[3] = d0[3];
            b[4] = d1[0]; b[5] = d1[1]; b[6] = d1[2]; b[7] = d1[3];
            bfrag[ni] = b;
        }

#pragma unroll
        for (int mi = 0; mi < 4; ++mi)
#pragma unroll
            for (int ni = 0; ni < 2; ++ni)
                acc[mi][ni] = __builtin_amdgcn_wmma_i32_16x16x64_iu8(
                     true, afrag[mi],
                     true, bfrag[ni],
                    acc[mi][ni],
                     false,  false);
    }

    const float si = scale_i[0];
#pragma unroll
    for (int ni = 0; ni < 2; ++ni) {
        const int   col = n_base + ni * 16 + l16;
        const float sc  = si * scale_w[col];
        const float bs  = bias[col];
#pragma unroll
        for (int mi = 0; mi < 4; ++mi) {
#pragma unroll
            for (int r = 0; r < 8; ++r) {
                const int row = m_base + mi * 16 + (half << 3) + r;
                out[(size_t)row * N_DIM + col] =
                    (float)acc[mi][ni][r] * sc + bs;
            }
        }
    }
}

extern "C" void kernel_launch(void* const* d_in, const int* in_sizes, int n_in,
                              void* d_out, int out_size, void* d_ws, size_t ws_size,
                              hipStream_t stream) {
    const int8_t* X  = (const int8_t*)d_in[0];
    const int8_t* Wq = (const int8_t*)d_in[1];
    const float*  si = (const float*)d_in[2];
    const float*  sw = (const float*)d_in[3];
    const float*  bs = (const float*)d_in[4];
    float*        o  = (float*)d_out;

    dim3 grid(N_DIM / TILE_MN, M_DIM / TILE_MN);
    int8_gemm_wmma_tdm<<<grid, dim3(256), 0, stream>>>(X, Wq, si, sw, bs, o);
}
